// RNN_64252710748536
// MI455X (gfx1250) — matrix operand order repaired and checked by reading, not compiled, not run on hardware
//
#include <hip/hip_runtime.h>
#include <hip/hip_bf16.h>

// ---------------------------------------------------------------------------
// LSTM LM forward: emb gather -> sequential LSTM scan (WMMA bf16 gates) ->
// batched emit GEMM [8192x32000, K=1024] fused with streaming log-softmax.
// Target: gfx1250 (CDNA5), wave32, v_wmma_f32_16x16x32_bf16.
// Emit GEMM is register-blocked 2(M-tiles) x 2(N-tiles) per wave iteration:
// 1 ds_load_b128 + 1 global_load_b128 per WMMA, halving L2 + LDS traffic
// vs the naive tiling (32-row blocks -> emit_w read 256x instead of 512x).
// ---------------------------------------------------------------------------

typedef __attribute__((ext_vector_type(16))) __bf16       v16bf;
typedef __attribute__((ext_vector_type(8)))  float        v8f;
typedef __attribute__((ext_vector_type(4)))  unsigned int v4u;

union ATile { v16bf v; v4u u[2]; };

#define B_ 32
#define T_ 256
#define E_ 1024
#define H_ 1024
#define V_ 32000
#define NEG_INF (-3.0e+38f)

// ---------------- prep kernels ---------------------------------------------

__global__ void gather_embed_bf16(const int* __restrict__ words,
                                  const float* __restrict__ embed_w,
                                  __bf16* __restrict__ xbf, int n) {
  // x layout: [B][T][E]; flat index i -> bt = i>>10, e = i&1023 (E=1024)
  for (int i = blockIdx.x * blockDim.x + threadIdx.x; i < n;
       i += gridDim.x * blockDim.x) {
    int bt = i >> 10, e = i & 1023;
    xbf[i] = (__bf16)embed_w[(size_t)words[bt] * E_ + e];
  }
}

__global__ void cvt_f32_bf16(const float* __restrict__ s,
                             __bf16* __restrict__ d, int n) {
  for (int i = blockIdx.x * blockDim.x + threadIdx.x; i < n;
       i += gridDim.x * blockDim.x)
    d[i] = (__bf16)s[i];
}

__global__ void init_state(__bf16* __restrict__ h0, float* __restrict__ c, int n) {
  for (int i = blockIdx.x * blockDim.x + threadIdx.x; i < n;
       i += gridDim.x * blockDim.x) {
    h0[i] = (__bf16)0.0f;
    c[i]  = 0.0f;
  }
}

// ---------------- LSTM step -------------------------------------------------
// gates[32,4096] = x_t[32,1024] @ w_ih^T + h[32,1024] @ w_hh^T + b
// One wave per (mtile in 0..1, jtile in 0..63): computes i/f/g/o tiles for the
// same 16 j-columns so the cell nonlinearity fuses on the accumulators.
// grid = 32 blocks x 128 threads (4 waves) = 128 waves.

__global__ void __launch_bounds__(128)
lstm_step(const __bf16* __restrict__ xbf,   // [B,T,E]
          const __bf16* __restrict__ wih,   // [4H,E] bf16
          const __bf16* __restrict__ whh,   // [4H,H] bf16
          const float* __restrict__ bih,
          const float* __restrict__ bhh,
          const __bf16* __restrict__ hin,   // [B,H]
          __bf16* __restrict__ hout,        // [B,H]
          float* __restrict__ c,            // [B,H]
          __bf16* __restrict__ hall,        // [T*B,H]
          int t) {
  const int lane  = threadIdx.x & 31;
  const int wid   = blockIdx.x * 4 + (threadIdx.x >> 5); // 0..127
  const int mtile = wid >> 6;                            // 0..1
  const int jtile = wid & 63;                            // 0..63
  const int rlo   = lane & 15;
  const int half  = lane >> 4;
  const int akb   = half * 8;   // A: K base within 0..31 (ISA A layout)
  const int bkb   = half * 8;   // B: K base within 0..31 (same table order as A)

  // A rows: batch index for this lane (A tile is 16xK, row = lane&15)
  const int brow = mtile * 16 + rlo;
  const __bf16* ax_base = xbf + ((size_t)brow * T_ + t) * E_;
  const __bf16* ah_base = hin + (size_t)brow * H_;

  // B rows: gate output column n = g*1024 + jtile*16 + (lane&15)
  const __bf16* bwx[4];
  const __bf16* bwh[4];
#pragma unroll
  for (int g = 0; g < 4; ++g) {
    int n = g * H_ + jtile * 16 + rlo;
    bwx[g] = wih + (size_t)n * E_ + bkb;
    bwh[g] = whh + (size_t)n * H_ + bkb;
  }

  v8f acc[4] = {};
  for (int kt = 0; kt < 32; ++kt) {
    const int k0 = kt * 32;
    ATile ax, ah;
    ax.u[0] = *(const v4u*)(ax_base + k0 + akb);
    ax.u[1] = *(const v4u*)(ax_base + k0 + akb + 16);
    ah.u[0] = *(const v4u*)(ah_base + k0 + akb);
    ah.u[1] = *(const v4u*)(ah_base + k0 + akb + 16);
#pragma unroll
    for (int g = 0; g < 4; ++g) {
      ATile bx, bh;
      bx.u[0] = *(const v4u*)(bwx[g] + k0);
      bx.u[1] = *(const v4u*)(bwx[g] + k0 + 16);
      bh.u[0] = *(const v4u*)(bwh[g] + k0);
      bh.u[1] = *(const v4u*)(bwh[g] + k0 + 16);
      acc[g] = __builtin_amdgcn_wmma_f32_16x16x32_bf16(
          false, ax.v, false, bx.v, (short)0, acc[g], false, false);
      acc[g] = __builtin_amdgcn_wmma_f32_16x16x32_bf16(
          false, ah.v, false, bh.v, (short)0, acc[g], false, false);
    }
  }

  // fused LSTM cell: C/D layout -> lane holds column jg, rows m = r + 8*half
  const int jg = jtile * 16 + rlo;
  float bias[4];
#pragma unroll
  for (int g = 0; g < 4; ++g) bias[g] = bih[g * H_ + jg] + bhh[g * H_ + jg];

#pragma unroll
  for (int r = 0; r < 8; ++r) {
    const int m = r + half * 8;
    const int b = mtile * 16 + m;
    float iv = acc[0][r] + bias[0];
    float fv = acc[1][r] + bias[1];
    float gv = acc[2][r] + bias[2];
    float ov = acc[3][r] + bias[3];
    iv = 1.0f / (1.0f + __expf(-iv));
    fv = 1.0f / (1.0f + __expf(-fv));
    ov = 1.0f / (1.0f + __expf(-ov));
    gv = tanhf(gv);
    const size_t idx = (size_t)b * H_ + jg;
    const float cn = fv * c[idx] + iv * gv;
    c[idx] = cn;
    const float hv = ov * tanhf(cn);
    const __bf16 hb = (__bf16)hv;
    hout[idx] = hb;
    hall[((size_t)t * B_ + b) * H_ + jg] = hb;
  }
}

// ---------------- emit GEMM + streaming log-softmax -------------------------
// One block per 32-row tile of Hall[8192,1024] (256 blocks, 8 waves).
// A (32x1024 bf16 = 64KB+pad) staged in dynamic LDS, padded stride 1032 bf16
// for conflict-free ds_load_b128. Each wave iterates over pairs of N-tiles
// (125 pairs/wave, uniform -> EXEC always full at WMMA) with a 2x2 register
// block: A tiles reused across 2 N-tiles, B tiles reused across 2 M-tiles.

#define LSTRIDE 1032
#define RPB 32              // rows per block
#define EMIT_LDS_BYTES (RPB * LSTRIDE * 2)

__global__ void __launch_bounds__(256)
emit_logprob(const __bf16* __restrict__ hall,  // [T*B, H]
             const __bf16* __restrict__ ew,    // [V, H] bf16
             const float* __restrict__ eb,     // [V]
             const int* __restrict__ words,    // [B, T]
             float* __restrict__ out) {        // [T*B] time-major
  extern __shared__ __align__(16) char smem_raw[];
  __bf16* la = (__bf16*)smem_raw;              // [RPB][LSTRIDE]
  __shared__ float redM[8][RPB];
  __shared__ float redS[8][RPB];
  __shared__ float tgt[RPB];

  const int tid     = threadIdx.x;
  const int lane    = tid & 31;
  const int wv      = tid >> 5;      // 0..7
  const int rowTile = blockIdx.x;    // 0..255

  // stage 32 rows x 1024 bf16 of Hall into LDS (16B chunks, 16 iters)
  for (int ch = tid; ch < RPB * 128; ch += 256) {
    const int r = ch >> 7, c8 = ch & 127;
    *(v4u*)(la + r * LSTRIDE + c8 * 8) =
        *(const v4u*)(hall + ((size_t)rowTile * RPB + r) * H_ + c8 * 8);
  }
  __syncthreads();

  const int rlo  = lane & 15;
  const int half = lane >> 4;
  const int akb  = half * 8;
  const int bkb  = half * 8;

  // target word per row this lane covers: rows mrow = r + 8*half + 16*mt
  int wtg[16];
#pragma unroll
  for (int mt = 0; mt < 2; ++mt) {
#pragma unroll
    for (int r = 0; r < 8; ++r) {
      const int R  = rowTile * RPB + (r + half * 8 + 16 * mt);  // = t*32 + b
      const int tt = R >> 5, bb = R & 31;
      wtg[mt * 8 + r] = words[bb * T_ + tt];
    }
  }

  float mx[16], sm[16];
#pragma unroll
  for (int i = 0; i < 16; ++i) { mx[i] = NEG_INF; sm[i] = 0.0f; }

  // 2000 N-tiles -> 1000 pairs, 125 pairs per wave (uniform, no divergence)
  for (int p = wv; p < (V_ / 32); p += 8) {
    const int v0 = p * 32;
    const __bf16* bp0 = ew + (size_t)(v0 + rlo) * H_ + bkb;
    const __bf16* bp1 = bp0 + (size_t)16 * H_;
    __builtin_prefetch(bp0 + (size_t)256 * H_, 0, 1);  // this wave's next pair
    v8f acc[2][2] = {};
#pragma unroll
    for (int kt = 0; kt < 32; ++kt) {
      ATile a0, a1, b0, b1;
      const __bf16* lp = la + rlo * LSTRIDE + kt * 32 + akb;
      a0.u[0] = *(const v4u*)lp;
      a0.u[1] = *(const v4u*)(lp + 16);
      a1.u[0] = *(const v4u*)(lp + 16 * LSTRIDE);
      a1.u[1] = *(const v4u*)(lp + 16 * LSTRIDE + 16);
      b0.u[0] = *(const v4u*)(bp0 + kt * 32);
      b0.u[1] = *(const v4u*)(bp0 + kt * 32 + 16);
      b1.u[0] = *(const v4u*)(bp1 + kt * 32);
      b1.u[1] = *(const v4u*)(bp1 + kt * 32 + 16);
      acc[0][0] = __builtin_amdgcn_wmma_f32_16x16x32_bf16(
          false, a0.v, false, b0.v, (short)0, acc[0][0], false, false);
      acc[0][1] = __builtin_amdgcn_wmma_f32_16x16x32_bf16(
          false, a0.v, false, b1.v, (short)0, acc[0][1], false, false);
      acc[1][0] = __builtin_amdgcn_wmma_f32_16x16x32_bf16(
          false, a1.v, false, b0.v, (short)0, acc[1][0], false, false);
      acc[1][1] = __builtin_amdgcn_wmma_f32_16x16x32_bf16(
          false, a1.v, false, b1.v, (short)0, acc[1][1], false, false);
    }
    // streaming logsumexp update + target-logit extraction
#pragma unroll
    for (int j = 0; j < 2; ++j) {
      const int col   = v0 + j * 16 + rlo;
      const float bias = eb[col];
#pragma unroll
      for (int mt = 0; mt < 2; ++mt) {
#pragma unroll
        for (int r = 0; r < 8; ++r) {
          const int i = mt * 8 + r;
          const float val = acc[mt][j][r] + bias;
          if (val > mx[i]) {
            sm[i] = sm[i] * __expf(mx[i] - val) + 1.0f;
            mx[i] = val;
          } else {
            sm[i] += __expf(val - mx[i]);
          }
          if (col == wtg[i]) tgt[r + half * 8 + 16 * mt] = val;
        }
      }
    }
  }

  // reduce (mx, sm) across the 16 lanes of each half (tile columns)
#pragma unroll
  for (int mt = 0; mt < 2; ++mt) {
#pragma unroll
    for (int r = 0; r < 8; ++r) {
      const int i = mt * 8 + r;
      float m = mx[i], s = sm[i];
#pragma unroll
      for (int off = 1; off < 16; off <<= 1) {
        const float mo = __shfl_xor(m, off);
        const float so = __shfl_xor(s, off);
        const float m2 = fmaxf(m, mo);
        s = s * __expf(m - m2) + so * __expf(mo - m2);
        m = m2;
      }
      if (rlo == 0) {
        redM[wv][r + half * 8 + 16 * mt] = m;
        redS[wv][r + half * 8 + 16 * mt] = s;
      }
    }
  }
  __syncthreads();

  // combine the 8 waves; emit ll = target_logit - (max + log(sumexp))
  if (tid < RPB) {
    float m = redM[0][tid], s = redS[0][tid];
#pragma unroll
    for (int w = 1; w < 8; ++w) {
      const float mo = redM[w][tid], so = redS[w][tid];
      const float m2 = fmaxf(m, mo);
      s = s * __expf(m - m2) + so * __expf(mo - m2);
      m = m2;
    }
    out[rowTile * RPB + tid] = tgt[tid] - (m + __logf(s));
  }
}

// ---------------- host launch ----------------------------------------------

extern "C" void kernel_launch(void* const* d_in, const int* in_sizes, int n_in,
                              void* d_out, int out_size, void* d_ws,
                              size_t ws_size, hipStream_t stream) {
  const int*   words   = (const int*)d_in[0];
  const float* embed_w = (const float*)d_in[1];
  const float* w_ih    = (const float*)d_in[2];
  const float* w_hh    = (const float*)d_in[3];
  const float* b_ih    = (const float*)d_in[4];
  const float* b_hh    = (const float*)d_in[5];
  const float* emit_w  = (const float*)d_in[6];
  const float* emit_b  = (const float*)d_in[7];
  float* out = (float*)d_out;

  char* ws = (char*)d_ws;
  size_t off = 0;
  auto take = [&](size_t bytes) -> char* {
    char* p = ws + off;
    off += (bytes + 255) & ~(size_t)255;
    return p;
  };
  __bf16* xbf  = (__bf16*)take((size_t)B_ * T_ * E_ * 2);  // 16 MB
  __bf16* wihb = (__bf16*)take((size_t)4 * H_ * E_ * 2);   //  8 MB
  __bf16* whhb = (__bf16*)take((size_t)4 * H_ * H_ * 2);   //  8 MB
  __bf16* ewb  = (__bf16*)take((size_t)V_ * H_ * 2);       // 64 MB
  __bf16* hall = (__bf16*)take((size_t)T_ * B_ * H_ * 2);  // 16 MB
  __bf16* hbuf = (__bf16*)take((size_t)2 * B_ * H_ * 2);
  float*  cbuf = (float*)take((size_t)B_ * H_ * 4);
  (void)ws_size; (void)in_sizes; (void)n_in; (void)out_size;

  gather_embed_bf16<<<4096, 256, 0, stream>>>(words, embed_w, xbf, B_ * T_ * E_);
  cvt_f32_bf16<<<2048, 256, 0, stream>>>(w_ih, wihb, 4 * H_ * E_);
  cvt_f32_bf16<<<2048, 256, 0, stream>>>(w_hh, whhb, 4 * H_ * H_);
  cvt_f32_bf16<<<8192, 256, 0, stream>>>(emit_w, ewb, V_ * H_);
  init_state<<<128, 256, 0, stream>>>(hbuf, cbuf, B_ * H_);

  for (int t = 0; t < T_; ++t) {
    __bf16* hin  = hbuf + (size_t)(t & 1) * (B_ * H_);
    __bf16* hout = hbuf + (size_t)((t + 1) & 1) * (B_ * H_);
    lstm_step<<<32, 128, 0, stream>>>(xbf, wihb, whhb, b_ih, b_hh, hin, hout,
                                      cbuf, hall, t);
  }

  emit_logprob<<<256, 256, EMIT_LDS_BYTES, stream>>>(hall, ewb, emit_b, words,
                                                     out);
}
